// Attention_16286515987100
// MI455X (gfx1250) — matrix operand order repaired and checked by reading, not compiled, not run on hardware
//
#include <hip/hip_runtime.h>

// ---------------- types ----------------
typedef _Float16 h16v __attribute__((ext_vector_type(16)));
typedef _Float16 h8v  __attribute__((ext_vector_type(8)));
typedef _Float16 h4v  __attribute__((ext_vector_type(4)));
typedef float    f8v  __attribute__((ext_vector_type(8)));
typedef float    f4v  __attribute__((ext_vector_type(4)));

#define HEADS 8
#define BATCH 8
#define NTOK  1024
#define DIMM  512
#define DHEAD 64
#define HB    (HEADS*BATCH)
#define EPSC  1e-6f

// ---------------- CDNA5 Tensor Data Mover (guarded) ----------------
#if defined(__has_builtin)
#if __has_builtin(__builtin_amdgcn_tensor_load_to_lds) && __has_builtin(__builtin_amdgcn_s_wait_tensorcnt)
#define ATTN_TDM 1
#endif
#endif
#ifndef ATTN_TDM
#define ATTN_TDM 0
#endif

#if ATTN_TDM
typedef unsigned int u32x4 __attribute__((ext_vector_type(4)));
typedef int          i32x4 __attribute__((ext_vector_type(4)));
typedef int          i32x8 __attribute__((ext_vector_type(8)));

// Issue a 2D TDM load: tile_y rows x tile_x f32 elements, row stride
// stride_elems, from global gsrc into LDS at lds_dst (dense row-major).
// D# packing per CDNA5 ISA 8.3/8.4. Tensor dims are 1<<20 so the OOB
// clamp never triggers for our in-bounds tiles.
__device__ __forceinline__ void tdm_load_tile_f32(const float* gsrc, void* lds_dst,
                                                  int tile_x, int tile_y,
                                                  int stride_elems) {
    unsigned long long ga = (unsigned long long)(uintptr_t)gsrc;
    u32x4 g0;
    g0[0] = 1u;                                               // count=1 (valid), user mode
    g0[1] = (unsigned)(uintptr_t)lds_dst;                     // lds_addr (bytes)
    g0[2] = (unsigned)ga;                                     // global_addr[31:0]
    g0[3] = (unsigned)((ga >> 32) & 0x01ffffffu) | (2u << 30);// addr[56:32] | type=2
    i32x8 g1;
    g1[0] = 2 << 16;                    // data_size = 4 bytes
    g1[1] = 0;                          // tensor_dim0 low16 = 0  (dim0 = 1<<20)
    g1[2] = 0x0010;                     // tensor_dim0 hi16 ; tensor_dim1 low16 = 0
    g1[3] = 0x0010 | (tile_x << 16);    // tensor_dim1 hi16 (1<<20) ; tile_dim0
    g1[4] = tile_y;                     // tile_dim1 ; tile_dim2 = 0
    g1[5] = stride_elems;               // tensor_dim0_stride low32
    g1[6] = 0;
    g1[7] = 0;
    i32x4 gz = {};                      // groups 2/3 unused for 2D tensors
#if __has_include(<hip/amd_detail/amd_gfx1250_TDM.h>)
    i32x8 gz8 = {};
    __builtin_amdgcn_tensor_load_to_lds(g0, g1, gz, gz, gz8, 0);
#else
    __builtin_amdgcn_tensor_load_to_lds(g0, g1, gz, gz, 0);
#endif
}
#endif

__device__ __forceinline__ f8v wmma_f16(h16v a, h16v b, f8v c) {
    return __builtin_amdgcn_wmma_f32_16x16x32_f16(false, a, false, b, (short)0, c, false, false);
}

__device__ __forceinline__ h16v load_frag(const _Float16* p0, const _Float16* p1) {
    h8v lo = *(const h8v*)p0;
    h8v hi = *(const h8v*)p1;
    h16v r;
#pragma unroll
    for (int i = 0; i < 8; ++i) { r[i] = lo[i]; r[i + 8] = hi[i]; }
    return r;
}

__device__ __forceinline__ float redsum16(float x) {
    x += __shfl_xor(x, 1); x += __shfl_xor(x, 2);
    x += __shfl_xor(x, 4); x += __shfl_xor(x, 8);
    return x;
}
__device__ __forceinline__ float redmax16(float x) {
    x = fmaxf(x, __shfl_xor(x, 1)); x = fmaxf(x, __shfl_xor(x, 2));
    x = fmaxf(x, __shfl_xor(x, 4)); x = fmaxf(x, __shfl_xor(x, 8));
    return x;
}
__device__ __forceinline__ float clip95(float x) { return fminf(fmaxf(x, -0.95f), 0.95f); }
__device__ __forceinline__ float clip20(float x) { return fminf(fmaxf(x, -20.f), 20.f); }

// ---------------- zero helper ----------------
__global__ void k_zero(float* p, int n) {
    int i = blockIdx.x * blockDim.x + threadIdx.x;
    if (i < n) p[i] = 0.f;
}

// ---------------- fused LayerNorm + input projection GEMM ----------------
// MODE 0: dst[((h*8+b)*1024+n)*64+dd]   (f_q / f_k row-major per (h,b))
// MODE 1: dst[((h*8+b)*64+dd)*1024+n]   (f_v transposed)
template <int MODE>
__global__ __launch_bounds__(128, 1)
void k_proj(const float* __restrict__ x, const float* __restrict__ lng,
            const float* __restrict__ lnb, const float* __restrict__ W,
            _Float16* __restrict__ dst) {
    const int tid = threadIdx.x;
    const int w = tid >> 5, lane = tid & 31, ln = lane & 15, hi = lane >> 4;
    const int Rbase = blockIdx.x * 64;
    const int hcol  = blockIdx.y;
    const int jc0   = hcol * 64;

    __shared__ float pS[64][2], pQ[64][2];
    __shared__ float mu[64], rstd[64];
    __shared__ __attribute__((aligned(16))) _Float16 At[64][48];
    __shared__ __attribute__((aligned(16))) _Float16 Bt[64][48];
    __shared__ __attribute__((aligned(16))) _Float16 Ct[64][72];
#if ATTN_TDM
    __shared__ __attribute__((aligned(16))) float Wstage[2][32][64];
    if (w == 0)   // prefetch K-tile 0 via Tensor Data Mover
        tdm_load_tile_f32(W + jc0, &Wstage[0][0][0], 64, 32, DIMM);
#endif

    { // row LN stats (2 threads per row, vectorized)
        int r = tid >> 1, half = tid & 1;
        const float* xr = x + (size_t)(Rbase + r) * DIMM + half * 256;
        float s = 0.f, q = 0.f;
        for (int c = 0; c < 256; c += 4) {
            f4v v4 = *(const f4v*)(xr + c);
#pragma unroll
            for (int e = 0; e < 4; ++e) { s += v4[e]; q += v4[e] * v4[e]; }
        }
        pS[r][half] = s; pQ[r][half] = q;
    }
    __syncthreads();
    if (tid < 64) {
        float s = pS[tid][0] + pS[tid][1];
        float q = pQ[tid][0] + pQ[tid][1];
        float m = s * (1.f / 512.f);
        mu[tid] = m;
        rstd[tid] = rsqrtf(q * (1.f / 512.f) - m * m + 1e-5f);
    }
    __syncthreads();

    f8v acc[4];
#pragma unroll
    for (int t = 0; t < 4; ++t) { f8v z = {}; acc[t] = z; }

    for (int kt = 0; kt < 16; ++kt) {
        int kk0 = kt * 32;
#if ATTN_TDM
        if (w == 0) __builtin_amdgcn_s_wait_tensorcnt(0);
        __syncthreads();                       // tile kt landed in Wstage[kt&1]
        if (w == 0 && kt + 1 < 16)             // async-prefetch tile kt+1
            tdm_load_tile_f32(W + (size_t)(kk0 + 32) * DIMM + jc0,
                              &Wstage[(kt + 1) & 1][0][0], 64, 32, DIMM);
#pragma unroll
        for (int i = 0; i < 4; ++i) {
            int f = tid + 128 * i;
            int j4 = (f & 15) * 4, kb = f >> 4;
            f4v v4 = *(const f4v*)&Wstage[kt & 1][kb][j4];
            Bt[j4 + 0][kb] = (_Float16)v4[0];
            Bt[j4 + 1][kb] = (_Float16)v4[1];
            Bt[j4 + 2][kb] = (_Float16)v4[2];
            Bt[j4 + 3][kb] = (_Float16)v4[3];
        }
#else
#pragma unroll
        for (int i = 0; i < 4; ++i) {
            int f = tid + 128 * i;
            int j4 = (f & 15) * 4, kb = f >> 4;
            f4v v4 = *(const f4v*)(W + (size_t)(kk0 + kb) * DIMM + jc0 + j4);
            Bt[j4 + 0][kb] = (_Float16)v4[0];
            Bt[j4 + 1][kb] = (_Float16)v4[1];
            Bt[j4 + 2][kb] = (_Float16)v4[2];
            Bt[j4 + 3][kb] = (_Float16)v4[3];
        }
#endif
#pragma unroll
        for (int i = 0; i < 4; ++i) {
            int f = tid + 128 * i;
            int r = f >> 3, k4 = (f & 7) * 4;
            f4v v4 = *(const f4v*)(x + (size_t)(Rbase + r) * DIMM + kk0 + k4);
            h4v o;
#pragma unroll
            for (int e = 0; e < 4; ++e)
                o[e] = (_Float16)((v4[e] - mu[r]) * rstd[r] * lng[kk0 + k4 + e] + lnb[kk0 + k4 + e]);
            *(h4v*)&At[r][k4] = o;
        }
        __syncthreads();
        int arow = 16 * w + ln;
        h16v af = load_frag(&At[arow][8 * hi], &At[arow][8 * hi + 16]);
#pragma unroll
        for (int t = 0; t < 4; ++t) {
            int j = 16 * t + ln;
            h16v bf = load_frag(&Bt[j][8 * hi], &Bt[j][8 * hi + 16]);
            acc[t] = wmma_f16(af, bf, acc[t]);
        }
        __syncthreads();
    }

#pragma unroll
    for (int t = 0; t < 4; ++t) {
        int col = 16 * t + ln;
#pragma unroll
        for (int r = 0; r < 8; ++r) {
            int lr = 16 * w + r + 8 * hi;
            _Float16 hv = (_Float16)acc[t][r];
            if (MODE == 0) Ct[lr][col] = hv;
            else           Ct[col][lr] = hv;
        }
    }
    __syncthreads();
#pragma unroll
    for (int i = 0; i < 4; ++i) {
        int f = tid + 128 * i;
        int a = f >> 3, ch = f & 7;
        if (MODE == 0) {
            int R = Rbase + a, b = R >> 10, n = R & 1023;
            *(h8v*)(dst + ((size_t)(hcol * BATCH + b) * NTOK + n) * DHEAD + ch * 8) =
                *(const h8v*)&Ct[a][ch * 8];
        } else {
            int nb = Rbase + ch * 8;
            int b = nb >> 10, n = nb & 1023;
            *(h8v*)(dst + ((size_t)(hcol * BATCH + b) * DHEAD + a) * NTOK + n) =
                *(const h8v*)&Ct[a][ch * 8];
        }
    }
}

// ---------------- per-(h,b) auxiliary vectors (incl. precomputed reciprocals) ----------------
__global__ __launch_bounds__(256, 1)
void k_aux(const _Float16* __restrict__ fq, const _Float16* __restrict__ fk,
           float* __restrict__ iqcb, float* __restrict__ iqmb,
           float* __restrict__ aqb, float* __restrict__ uqb,
           float* __restrict__ ikcb, float* __restrict__ ikmb,
           float* __restrict__ skb, float* __restrict__ Cb,
           float* __restrict__ qg_acc, float* __restrict__ kg_acc) {
    int hb = blockIdx.x, tid = threadIdx.x;
    int h = hb >> 3;
    const _Float16* K = fk + (size_t)hb * NTOK * DHEAD;
    const _Float16* Q = fq + (size_t)hb * NTOK * DHEAD;
    __shared__ float colp[4][64];
    __shared__ float kb[64];
    __shared__ float Cred[64];

    for (int rr = 0; rr < 4; ++rr) {
        int m = tid + rr * 256;
        const _Float16* row = K + (size_t)m * DHEAD;
        float s = 0.f, q = 0.f;
        for (int c = 0; c < DHEAD; ++c) { float v = (float)row[c]; s += v; q += v * v; }
        float nk = sqrtf(q);
        skb [(size_t)hb * NTOK + m] = s;
        ikcb[(size_t)hb * NTOK + m] = 1.f / (nk + EPSC);
        ikmb[(size_t)hb * NTOK + m] = 1.f / fmaxf(nk, EPSC);
    }
    int c = tid & 63, g = tid >> 6;
    float cs = 0.f;
    for (int m = g * 256; m < g * 256 + 256; ++m) cs += (float)K[(size_t)m * DHEAD + c];
    colp[g][c] = cs;
    __syncthreads();
    if (tid < 64) {
        float tot = colp[0][tid] + colp[1][tid] + colp[2][tid] + colp[3][tid];
        kb[tid] = tot * (1.f / 1024.f);
        Cred[tid] = kb[tid];
        atomicAdd(&kg_acc[h * 64 + tid], tot);
    }
    __syncthreads();
    for (int off = 32; off > 0; off >>= 1) {
        if (tid < off) Cred[tid] += Cred[tid + off];
        __syncthreads();
    }
    if (tid == 0) Cb[hb] = Cred[0];
    __syncthreads();
    for (int rr = 0; rr < 4; ++rr) {
        int n = tid + rr * 256;
        const _Float16* row = Q + (size_t)n * DHEAD;
        float s = 0.f, q = 0.f, uu = 0.f;
        for (int c2 = 0; c2 < DHEAD; ++c2) {
            float v = (float)row[c2];
            s += v; q += v * v; uu += v * kb[c2];
        }
        float nq = sqrtf(q);
        aqb [(size_t)hb * NTOK + n] = s * (1.f / 64.f);
        uqb [(size_t)hb * NTOK + n] = uu;
        iqcb[(size_t)hb * NTOK + n] = 1.f / (nq + EPSC);
        iqmb[(size_t)hb * NTOK + n] = 1.f / fmaxf(nq, EPSC);
    }
    float cq = 0.f;
    for (int m = g * 256; m < g * 256 + 256; ++m) cq += (float)Q[(size_t)m * DHEAD + c];
    __syncthreads();
    colp[g][c] = cq;
    __syncthreads();
    if (tid < 64) {
        float tot = colp[0][tid] + colp[1][tid] + colp[2][tid] + colp[3][tid];
        atomicAdd(&qg_acc[h * 64 + tid], tot);
    }
}

// ---------------- pass A: S = fq.fk^T tiles, accumulate per-row moments ----------------
__global__ __launch_bounds__(128, 1)
void k_passA(const _Float16* __restrict__ fq, const _Float16* __restrict__ fk,
             const float* __restrict__ iqcb, const float* __restrict__ iqmb,
             const float* __restrict__ aqb, const float* __restrict__ uqb,
             const float* __restrict__ ikcb, const float* __restrict__ ikmb,
             const float* __restrict__ skb, const float* __restrict__ Cb,
             float* __restrict__ rowagg) {
    const int hb = blockIdx.y;
    const int tid = threadIdx.x, w = tid >> 5, lane = tid & 31, ln = lane & 15, hi = lane >> 4;
    const int n0 = blockIdx.x * 64 + 16 * w;
    const size_t base = (size_t)hb * NTOK;
    const float COVS = 0.001f / (1024.f * (8.f + 1e-6f));

    const _Float16* qrow = fq + (base + n0 + ln) * DHEAD;
    h16v a0 = load_frag(qrow + 8 * hi,      qrow + 8 * hi + 16);
    h16v a1 = load_frag(qrow + 32 + 8 * hi, qrow + 32 + 8 * hi + 16);

    float Cc = Cb[hb];
    float iqc[8], iqm[8], a_[8], d_[8];
#pragma unroll
    for (int r = 0; r < 8; ++r) {
        int row = n0 + r + 8 * hi;
        iqc[r] = iqcb[base + row];
        iqm[r] = iqmb[base + row];
        a_[r]  = aqb[base + row];
        d_[r]  = uqb[base + row] - a_[r] * Cc;   // cov = (S - d - a*s)*COVS
    }

    float sc[8] = {}, sc2[8] = {}, sv[8] = {}, sv2[8] = {}, scv[8] = {}, smg[8] = {};

    for (int mt = 0; mt < 16; ++mt) {
        int mbase = mt * 64;
#pragma unroll
        for (int t = 0; t < 4; ++t) {
            int m = mbase + 16 * t + ln;
            const _Float16* krow = fk + (base + m) * DHEAD;
            __builtin_prefetch(krow + 64 * DHEAD, 0, 1);
            h16v b0 = load_frag(krow + 8 * hi,      krow + 8 * hi + 16);
            h16v b1 = load_frag(krow + 32 + 8 * hi, krow + 32 + 8 * hi + 16);
            f8v s = {};
            s = wmma_f16(a0, b0, s);
            s = wmma_f16(a1, b1, s);
            float ikc = ikcb[base + m];
            float ikm = ikmb[base + m];
            float sk_ = skb[base + m];
#pragma unroll
            for (int r = 0; r < 8; ++r) {
                float Sv = s[r];
                float cosv = clip95(Sv * iqc[r] * ikc);
                float csv  = clip95(Sv * iqm[r] * ikm);
                float covv = clip20((Sv - d_[r] - a_[r] * sk_) * COVS);
                float mg = fminf(fmaxf(0.01f - csv, 0.f), 5.f);
                sc[r] += cosv; sc2[r] += cosv * cosv;
                sv[r] += covv; sv2[r] += covv * covv;
                scv[r] += cosv * covv; smg[r] += mg;
            }
        }
    }
#pragma unroll
    for (int r = 0; r < 8; ++r) {
        float v0 = redsum16(sc[r]),  v1 = redsum16(sc2[r]), v2 = redsum16(sv[r]);
        float v3 = redsum16(sv2[r]), v4 = redsum16(scv[r]), v5 = redsum16(smg[r]);
        if (ln == 0) {
            float* d = rowagg + (base + n0 + r + 8 * hi) * 8;
            d[0] = v0; d[1] = v1; d[2] = v2; d[3] = v3; d[4] = v4; d[5] = v5;
        }
    }
}

// ---------------- weight-predictor MLP (one thread per head) ----------------
__global__ void k_mlp(const float* qg_acc, const float* kg_acc,
                      const float* W1, const float* b1, const float* lg, const float* lb,
                      const float* W2, const float* b2, const float* W3, const float* b3,
                      const float* wtemp, float* wout) {
    int h = threadIdx.x;
    if (h >= HEADS) return;
    float z1[64];
    for (int j = 0; j < 64; ++j) {
        float s = b1[j];
        for (int c = 0; c < 64; ++c) s += (qg_acc[h * 64 + c] * (1.f / 8192.f)) * W1[c * 64 + j];
        for (int c = 0; c < 64; ++c) s += (kg_acc[h * 64 + c] * (1.f / 8192.f)) * W1[(64 + c) * 64 + j];
        z1[j] = s;
    }
    float mu = 0.f;
    for (int j = 0; j < 64; ++j) mu += z1[j];
    mu *= (1.f / 64.f);
    float var = 0.f;
    for (int j = 0; j < 64; ++j) { float d = z1[j] - mu; var += d * d; }
    var *= (1.f / 64.f);
    float rstd = rsqrtf(var + 1e-5f);
    for (int j = 0; j < 64; ++j)
        z1[j] = fmaxf((z1[j] - mu) * rstd * lg[j] + lb[j], 0.f);
    float z2[32];
    for (int j = 0; j < 32; ++j) {
        float s = b2[j];
        for (int c = 0; c < 64; ++c) s += z1[c] * W2[c * 32 + j];
        z2[j] = fmaxf(s, 0.f);
    }
    float z3[3];
    for (int j = 0; j < 3; ++j) {
        float s = b3[j];
        for (int c = 0; c < 32; ++c) s += z2[c] * W3[c * 3 + j];
        z3[j] = s;
    }
    float m1 = fmaxf(z3[0], fmaxf(z3[1], z3[2]));
    float e0 = __expf(z3[0] - m1), e1 = __expf(z3[1] - m1), e2 = __expf(z3[2] - m1);
    float se = e0 + e1 + e2;
    float p0 = e0 / se, p1 = e1 / se, p2 = e2 / se;
    float wt = fminf(fmaxf(wtemp[0], 0.1f), 10.f);
    float q0 = p0 / wt, q1 = p1 / wt, q2 = p2 / wt;
    float m2 = fmaxf(q0, fmaxf(q1, q2));
    float f0 = __expf(q0 - m2), f1 = __expf(q1 - m2), f2 = __expf(q2 - m2);
    float sf = f0 + f1 + f2;
    wout[h * 3 + 0] = f0 / sf * 0.7f + 0.1f;
    wout[h * 3 + 1] = f1 / sf * 0.7f + 0.1f;
    wout[h * 3 + 2] = f2 / sf * 0.7f + 0.1f;
}

// ---------------- stats reduce: per-head 9 moment sums ----------------
__global__ __launch_bounds__(256, 1)
void k_s1(const float* __restrict__ rowagg, float* __restrict__ hsums) {
    int h = blockIdx.x, tid = threadIdx.x;
    float acc[9] = {};
    for (int idx = tid; idx < BATCH * NTOK; idx += 256) {
        const float* ra = rowagg + ((size_t)h * BATCH * NTOK + idx) * 8;
        float rc = ra[0], rc2 = ra[1], rv = ra[2], rv2 = ra[3], rcv = ra[4];
        float vr = ra[5] * (1.f / 1024.f);
        acc[0] += rc; acc[1] += rc2; acc[2] += rv; acc[3] += rv2; acc[4] += rcv;
        acc[5] += 1024.f * vr; acc[6] += 1024.f * vr * vr;
        acc[7] += vr * rc; acc[8] += vr * rv;
    }
    __shared__ float red[256];
    for (int j = 0; j < 9; ++j) {
        red[tid] = acc[j];
        __syncthreads();
        for (int off = 128; off > 0; off >>= 1) {
            if (tid < off) red[tid] += red[tid + off];
            __syncthreads();
        }
        if (tid == 0) hsums[h * 9 + j] = red[0];
        __syncthreads();
    }
}

// ---------------- final scalar combine: global stds, temp, coeffs ----------------
__global__ void k_s2(const float* hs, const float* w, float* coeffs) {
    if (threadIdx.x != 0 || blockIdx.x != 0) return;
    const double T = 64.0 * 1024.0 * 1024.0;
    double Sc = 0, Sc2 = 0, Sv = 0, Sv2 = 0, Sg = 0, Sg2 = 0;
    for (int h = 0; h < HEADS; ++h) {
        Sc += hs[h * 9 + 0]; Sc2 += hs[h * 9 + 1];
        Sv += hs[h * 9 + 2]; Sv2 += hs[h * 9 + 3];
        Sg += hs[h * 9 + 5]; Sg2 += hs[h * 9 + 6];
    }
    auto std1 = [&](double s, double q) {
        double v = (q - s * s / T) / (T - 1.0);
        return v > 0.0 ? sqrt(v) : 0.0;
    };
    double cos_n = std1(Sc, Sc2) + 1e-6;
    double cov_n = std1(Sv, Sv2) + 1e-6;
    double var_n = std1(Sg, Sg2) + 1e-6;
    double sd = 0, sd2 = 0;
    for (int h = 0; h < HEADS; ++h) {
        double al = (double)w[h * 3 + 0] / cos_n;
        double be = (double)w[h * 3 + 1] * 0.3 / cov_n;
        double ga = (double)w[h * 3 + 2] * 0.3 / var_n;
        sd  += al * hs[h * 9 + 0] + be * hs[h * 9 + 2] + ga * hs[h * 9 + 5];
        sd2 += al * al * hs[h * 9 + 1] + be * be * hs[h * 9 + 3] + ga * ga * hs[h * 9 + 6]
             + 2.0 * al * be * hs[h * 9 + 4] + 2.0 * al * ga * hs[h * 9 + 7]
             + 2.0 * be * ga * hs[h * 9 + 8];
    }
    double temp = 0.5 + std1(sd, sd2);
    temp = temp < 0.3 ? 0.3 : (temp > 3.0 ? 3.0 : temp);
    for (int h = 0; h < HEADS; ++h) {
        coeffs[h * 3 + 0] = (float)((double)w[h * 3 + 0] / cos_n / temp);
        coeffs[h * 3 + 1] = (float)((double)w[h * 3 + 1] * 0.3 / cov_n / temp);
        coeffs[h * 3 + 2] = (float)((double)w[h * 3 + 2] * 0.3 / var_n / temp);
    }
}

// ---------------- pass B: flash-attention with mixed dots ----------------
__global__ __launch_bounds__(128, 1)
void k_passB(const _Float16* __restrict__ fq, const _Float16* __restrict__ fk,
             const _Float16* __restrict__ fvT,
             const float* __restrict__ iqcb, const float* __restrict__ aqb,
             const float* __restrict__ uqb, const float* __restrict__ ikcb,
             const float* __restrict__ skb, const float* __restrict__ Cb,
             const float* __restrict__ rowagg, const float* __restrict__ coeffs,
             _Float16* __restrict__ Obuf) {
    const int hb = blockIdx.y, h = hb >> 3, bb = hb & 7;
    const int tid = threadIdx.x, w = tid >> 5, lane = tid & 31, ln = lane & 15, hi = lane >> 4;
    const int n0 = blockIdx.x * 64 + 16 * w;
    const size_t base = (size_t)hb * NTOK;
    const float COVS = 0.001f / (1024.f * (8.f + 1e-6f));

    __shared__ __attribute__((aligned(16))) _Float16 Pl[4][16][64];
    __shared__ __attribute__((aligned(16))) _Float16 Ot[64][72];

    const _Float16* qrow = fq + (base + n0 + ln) * DHEAD;
    h16v a0 = load_frag(qrow + 8 * hi,      qrow + 8 * hi + 16);
    h16v a1 = load_frag(qrow + 32 + 8 * hi, qrow + 32 + 8 * hi + 16);

    float Cc = Cb[hb];
    float iqc[8], a_[8], d_[8], varr[8], rmax[8], rsum[8];
#pragma unroll
    for (int r = 0; r < 8; ++r) {
        int row = n0 + r + 8 * hi;
        iqc[r] = iqcb[base + row];
        a_[r]  = aqb[base + row];
        d_[r]  = uqb[base + row] - a_[r] * Cc;
        varr[r] = rowagg[(base + row) * 8 + 5] * (1.f / 1024.f);
        rmax[r] = -3.0e38f;
        rsum[r] = 0.f;
    }
    float Ah = coeffs[h * 3 + 0], Bh = coeffs[h * 3 + 1], Gh = coeffs[h * 3 + 2];

    f8v O[4];
#pragma unroll
    for (int t = 0; t < 4; ++t) { f8v z = {}; O[t] = z; }

    for (int mt = 0; mt < 16; ++mt) {
        int mbase = mt * 64;
        f8v s4[4];
#pragma unroll
        for (int t = 0; t < 4; ++t) {
            int m = mbase + 16 * t + ln;
            const _Float16* krow = fk + (base + m) * DHEAD;
            __builtin_prefetch(krow + 64 * DHEAD, 0, 1);
            h16v b0 = load_frag(krow + 8 * hi,      krow + 8 * hi + 16);
            h16v b1 = load_frag(krow + 32 + 8 * hi, krow + 32 + 8 * hi + 16);
            f8v s = {};
            s = wmma_f16(a0, b0, s);
            s = wmma_f16(a1, b1, s);
            float ikc = ikcb[base + m];
            float sk_ = skb[base + m];
#pragma unroll
            for (int r = 0; r < 8; ++r) {
                float Sv = s[r];
                float cosv = clip95(Sv * iqc[r] * ikc);
                float covv = clip20((Sv - d_[r] - a_[r] * sk_) * COVS);
                s[r] = Ah * cosv + Bh * covv + Gh * varr[r];
            }
            s4[t] = s;
        }
        float scale[8], nmax[8];
#pragma unroll
        for (int r = 0; r < 8; ++r) {
            float tm = fmaxf(fmaxf(s4[0][r], s4[1][r]), fmaxf(s4[2][r], s4[3][r]));
            tm = redmax16(tm);
            float nm = fmaxf(rmax[r], tm);
            scale[r] = __expf(rmax[r] - nm);
            nmax[r] = nm;
            rmax[r] = nm;
        }
        float psum[8] = {};
#pragma unroll
        for (int t = 0; t < 4; ++t) {
#pragma unroll
            for (int r = 0; r < 8; ++r) {
                float p = __expf(s4[t][r] - nmax[r]);
                psum[r] += p;
                Pl[w][r + 8 * hi][16 * t + ln] = (_Float16)p;
            }
        }
#pragma unroll
        for (int r = 0; r < 8; ++r) {
            float tot = redsum16(psum[r]);
            rsum[r] = rsum[r] * scale[r] + tot;
        }
#pragma unroll
        for (int t = 0; t < 4; ++t)
#pragma unroll
            for (int r = 0; r < 8; ++r) O[t][r] *= scale[r];
        __syncthreads();
        const _Float16* prow = &Pl[w][ln][0];
        h16v p0 = load_frag(prow + 8 * hi,      prow + 8 * hi + 16);
        h16v p1 = load_frag(prow + 32 + 8 * hi, prow + 32 + 8 * hi + 16);
#pragma unroll
        for (int t = 0; t < 4; ++t) {
            int dd = 16 * t + ln;
            const _Float16* vrow = fvT + ((size_t)hb * DHEAD + dd) * NTOK + mbase;
            h16v b0 = load_frag(vrow + 8 * hi,      vrow + 8 * hi + 16);
            h16v b1 = load_frag(vrow + 32 + 8 * hi, vrow + 32 + 8 * hi + 16);
            O[t] = wmma_f16(p0, b0, O[t]);
            O[t] = wmma_f16(p1, b1, O[t]);
        }
        __syncthreads();
    }
#pragma unroll
    for (int t = 0; t < 4; ++t) {
#pragma unroll
        for (int r = 0; r < 8; ++r) {
            int lr = 16 * w + r + 8 * hi;
            Ot[lr][16 * t + ln] = (_Float16)(O[t][r] / rsum[r]);
        }
    }
    __syncthreads();
#pragma unroll
    for (int i = 0; i < 4; ++i) {
        int f = tid + 128 * i;
        int lr = f >> 3, ch = f & 7;
        int row = blockIdx.x * 64 + lr;
        *(h8v*)(Obuf + ((size_t)(bb * NTOK + row)) * DIMM + h * DHEAD + ch * 8) =
            *(const h8v*)&Ot[lr][ch * 8];
    }
}

// ---------------- output GEMM: Obuf[8192x512] @ W_out + b_out -> f32 ----------------
__global__ __launch_bounds__(128, 1)
void k_out(const _Float16* __restrict__ A, const float* __restrict__ W,
           const float* __restrict__ bias, float* __restrict__ out) {
    const int tid = threadIdx.x;
    const int w = tid >> 5, lane = tid & 31, ln = lane & 15, hi = lane >> 4;
    const int Rbase = blockIdx.x * 64;
    const int jc0 = blockIdx.y * 64;
    __shared__ __attribute__((aligned(16))) _Float16 Bt[64][48];
    __shared__ __attribute__((aligned(16))) float Cf[64][68];
#if ATTN_TDM
    __shared__ __attribute__((aligned(16))) float Wstage[2][32][64];
    if (w == 0)
        tdm_load_tile_f32(W + jc0, &Wstage[0][0][0], 64, 32, DIMM);
#endif

    f8v acc[4];
#pragma unroll
    for (int t = 0; t < 4; ++t) { f8v z = {}; acc[t] = z; }

    const _Float16* arow_p = A + (size_t)(Rbase + 16 * w + ln) * DIMM;
    for (int kt = 0; kt < 16; ++kt) {
        int kk0 = kt * 32;
#if ATTN_TDM
        if (w == 0) __builtin_amdgcn_s_wait_tensorcnt(0);
        __syncthreads();
        if (w == 0 && kt + 1 < 16)
            tdm_load_tile_f32(W + (size_t)(kk0 + 32) * DIMM + jc0,
                              &Wstage[(kt + 1) & 1][0][0], 64, 32, DIMM);
#pragma unroll
        for (int i = 0; i < 4; ++i) {
            int f = tid + 128 * i;
            int j4 = (f & 15) * 4, kb = f >> 4;
            f4v v4 = *(const f4v*)&Wstage[kt & 1][kb][j4];
            Bt[j4 + 0][kb] = (_Float16)v4[0];
            Bt[j4 + 1][kb] = (_Float16)v4[1];
            Bt[j4 + 2][kb] = (_Float16)v4[2];
            Bt[j4 + 3][kb] = (_Float16)v4[3];
        }
#else
#pragma unroll
        for (int i = 0; i < 4; ++i) {
            int f = tid + 128 * i;
            int j4 = (f & 15) * 4, kb = f >> 4;
            f4v v4 = *(const f4v*)(W + (size_t)(kk0 + kb) * DIMM + jc0 + j4);
            Bt[j4 + 0][kb] = (_Float16)v4[0];
            Bt[j4 + 1][kb] = (_Float16)v4[1];
            Bt[j4 + 2][kb] = (_Float16)v4[2];
            Bt[j4 + 3][kb] = (_Float16)v4[3];
        }
#endif
        __syncthreads();
        h16v af = load_frag(arow_p + kk0 + 8 * hi, arow_p + kk0 + 8 * hi + 16);
#pragma unroll
        for (int t = 0; t < 4; ++t) {
            int j = 16 * t + ln;
            h16v bf = load_frag(&Bt[j][8 * hi], &Bt[j][8 * hi + 16]);
            acc[t] = wmma_f16(af, bf, acc[t]);
        }
        __syncthreads();
    }
#pragma unroll
    for (int t = 0; t < 4; ++t) {
        int col = 16 * t + ln;
        float bv = bias[jc0 + col];
#pragma unroll
        for (int r = 0; r < 8; ++r) {
            int lr = 16 * w + r + 8 * hi;
            Cf[lr][col] = acc[t][r] + bv;
        }
    }
    __syncthreads();
#pragma unroll
    for (int i = 0; i < 8; ++i) {
        int f = tid + 128 * i;
        int lr = f >> 4, ch = f & 15;
        *(f4v*)(out + (size_t)(Rbase + lr) * DIMM + jc0 + ch * 4) =
            *(const f4v*)&Cf[lr][ch * 4];
    }
}

// ---------------- launcher ----------------
extern "C" void kernel_launch(void* const* d_in, const int* in_sizes, int n_in,
                              void* d_out, int out_size, void* d_ws, size_t ws_size,
                              hipStream_t stream) {
    const float* q     = (const float*)d_in[0];
    const float* k     = (const float*)d_in[1];
    const float* v     = (const float*)d_in[2];
    const float* ln_g  = (const float*)d_in[3];
    const float* ln_b  = (const float*)d_in[4];
    const float* W_in  = (const float*)d_in[5];
    const float* wp_W1 = (const float*)d_in[6];
    const float* wp_b1 = (const float*)d_in[7];
    const float* wp_lg = (const float*)d_in[8];
    const float* wp_lb = (const float*)d_in[9];
    const float* wp_W2 = (const float*)d_in[10];
    const float* wp_b2 = (const float*)d_in[11];
    const float* wp_W3 = (const float*)d_in[12];
    const float* wp_b3 = (const float*)d_in[13];
    const float* wtemp = (const float*)d_in[14];
    const float* W_out = (const float*)d_in[15];
    const float* b_out = (const float*)d_in[16];

    char* ws = (char*)d_ws;
    size_t off = 0;
    auto carve = [&](size_t bytes) -> char* {
        char* p = ws + off;
        off = (off + bytes + 255) & ~(size_t)255;
        return p;
    };
    const size_t NF = (size_t)HB * NTOK * DHEAD;   // 4,194,304 elems
    _Float16* fq   = (_Float16*)carve(NF * 2);
    _Float16* fk   = (_Float16*)carve(NF * 2);
    _Float16* fvT  = (_Float16*)carve(NF * 2);
    _Float16* Obuf = (_Float16*)carve((size_t)BATCH * NTOK * DIMM * 2);
    float* rowagg = (float*)carve((size_t)HB * NTOK * 8 * 4);
    float* iqcb = (float*)carve((size_t)HB * NTOK * 4);
    float* iqmb = (float*)carve((size_t)HB * NTOK * 4);
    float* aqb  = (float*)carve((size_t)HB * NTOK * 4);
    float* uqb  = (float*)carve((size_t)HB * NTOK * 4);
    float* ikcb = (float*)carve((size_t)HB * NTOK * 4);
    float* ikmb = (float*)carve((size_t)HB * NTOK * 4);
    float* skb  = (float*)carve((size_t)HB * NTOK * 4);
    float* Cb   = (float*)carve(HB * 4);
    float* qkg  = (float*)carve(1024 * 4);         // qg_acc[512] ++ kg_acc[512]
    float* qg_acc = qkg, * kg_acc = qkg + 512;
    float* wbuf   = (float*)carve(HEADS * 3 * 4);
    float* hsums  = (float*)carve(HEADS * 9 * 4);
    float* coeffs = (float*)carve(HEADS * 3 * 4);

    k_zero<<<4, 256, 0, stream>>>(qkg, 1024);
    k_proj<0><<<dim3(128, 8), 128, 0, stream>>>(q, ln_g, ln_b, W_in, fq);
    k_proj<0><<<dim3(128, 8), 128, 0, stream>>>(k, ln_g, ln_b, W_in, fk);
    k_proj<1><<<dim3(128, 8), 128, 0, stream>>>(v, ln_g, ln_b, W_in, fvT);
    k_aux<<<HB, 256, 0, stream>>>(fq, fk, iqcb, iqmb, aqb, uqb, ikcb, ikmb, skb, Cb,
                                  qg_acc, kg_acc);
    k_passA<<<dim3(16, HB), 128, 0, stream>>>(fq, fk, iqcb, iqmb, aqb, uqb, ikcb, ikmb,
                                              skb, Cb, rowagg);
    k_mlp<<<1, 32, 0, stream>>>(qg_acc, kg_acc, wp_W1, wp_b1, wp_lg, wp_lb,
                                wp_W2, wp_b2, wp_W3, wp_b3, wtemp, wbuf);
    k_s1<<<HEADS, 256, 0, stream>>>(rowagg, hsums);
    k_s2<<<1, 1, 0, stream>>>(hsums, wbuf, coeffs);
    k_passB<<<dim3(16, HB), 128, 0, stream>>>(fq, fk, fvT, iqcb, aqb, uqb, ikcb, skb, Cb,
                                              rowagg, coeffs, Obuf);
    k_out<<<dim3(128, 8), 128, 0, stream>>>(Obuf, W_out, b_out, (float*)d_out);
}
